// SwinTransformer_43009802502185
// MI455X (gfx1250) — compile-verified, matrix operand order checked by reading, not run on hardware
//
#include <hip/hip_runtime.h>
#include <math.h>

// ---------------------------------------------------------------------------
// Swin-T forward for MI455X (gfx1250, wave32).  All GEMMs (QKV/proj/MLP/merge/
// head) and both attention matmuls run on v_wmma_f32_16x16x32_f16.
// Fragment gathers are 16B-aligned ds_load_b128s (K-contiguous layouts only).
// GEMM tile: 128x64x32, 8 waves, 4 WMMA per wave per K-step.
// ---------------------------------------------------------------------------

typedef __attribute__((ext_vector_type(16))) _Float16 v16h;
typedef __attribute__((ext_vector_type(8)))  _Float16 v8h;
typedef __attribute__((ext_vector_type(8)))  float    v8f;
typedef __attribute__((ext_vector_type(4)))  float    v4f;

union FragC  { v8f v; float    f[8]; };
union Half8  { v8h v; _Float16 h[8]; };

// WMMA 16-bit A/B fragment (ISA 7.12.2): lane l, element e -> K =
// (l<16?0:8) + (e>>3)*16 + (e&7).  With a K-contiguous row in LDS this is two
// contiguous 8-half (16 B) runs at [base] and [base+16].
__device__ __forceinline__ v16h frag16(const _Float16* row, int base) {
    const v8h lo = *(const v8h*)(row + base);
    const v8h hi = *(const v8h*)(row + base + 16);
    return __builtin_shufflevector(lo, hi, 0, 1, 2, 3, 4, 5, 6, 7,
                                           8, 9, 10, 11, 12, 13, 14, 15);
}

__device__ __forceinline__ v8h pack8(v4f a, v4f b) {
    Half8 r;
    r.h[0] = (_Float16)a.x; r.h[1] = (_Float16)a.y;
    r.h[2] = (_Float16)a.z; r.h[3] = (_Float16)a.w;
    r.h[4] = (_Float16)b.x; r.h[5] = (_Float16)b.y;
    r.h[6] = (_Float16)b.z; r.h[7] = (_Float16)b.w;
    return r.v;
}

__device__ __forceinline__ v8h zero8() {
    Half8 z;
    #pragma unroll
    for (int e = 0; e < 8; ++e) z.h[e] = (_Float16)0.0f;
    return z.v;
}

// ---------------------------------------------------------------------------
// Patch embed: 4x4 stride-4 conv (NCHW in, NHWC out) + bias.
// ---------------------------------------------------------------------------
__global__ __launch_bounds__(256) void patch_embed_kernel(
    const float* __restrict__ x, const float* __restrict__ w,
    const float* __restrict__ bias, float* __restrict__ out, int total)
{
    int i = blockIdx.x * 256 + threadIdx.x;
    if (i >= total) return;
    int co = i % 96;
    int t  = i / 96;
    int ow = t % 56; t /= 56;
    int oh = t % 56;
    int b  = t / 56;
    float acc = bias[co];
    for (int ci = 0; ci < 3; ++ci)
        for (int kh = 0; kh < 4; ++kh)
            for (int kw = 0; kw < 4; ++kw)
                acc += x[((b * 3 + ci) * 224 + oh * 4 + kh) * 224 + ow * 4 + kw] *
                       w[((co * 3 + ci) * 4 + kh) * 4 + kw];
    out[i] = acc;
}

// ---------------------------------------------------------------------------
// Row-wise LayerNorm: one block per token row.
// ---------------------------------------------------------------------------
__global__ __launch_bounds__(256) void layernorm_kernel(
    const float* __restrict__ in, const float* __restrict__ g,
    const float* __restrict__ b, float* __restrict__ out, int C)
{
    __shared__ float rs[256], rs2[256];
    __shared__ float mean_s, rstd_s;
    int row = blockIdx.x;
    int tid = threadIdx.x;
    unsigned base = (unsigned)row * (unsigned)C;
    float s = 0.f, s2 = 0.f;
    for (int c = tid; c < C; c += 256) {
        float v = in[base + c];
        s += v; s2 += v * v;
    }
    rs[tid] = s; rs2[tid] = s2;
    __syncthreads();
    for (int off = 128; off > 0; off >>= 1) {
        if (tid < off) { rs[tid] += rs[tid + off]; rs2[tid] += rs2[tid + off]; }
        __syncthreads();
    }
    if (tid == 0) {
        float m = rs[0] / C;
        float v = rs2[0] / C - m * m;
        mean_s = m;
        rstd_s = rsqrtf(v + 1e-5f);
    }
    __syncthreads();
    float m = mean_s, r = rstd_s;
    for (int c = tid; c < C; c += 256)
        out[base + c] = (in[base + c] - m) * r * g[c] + b[c];
}

// ---------------------------------------------------------------------------
// WMMA GEMM: out[M,N] = epilogue(A[M,K] @ W[K,N]).  K % 32 == 0.
// mode: 0 = none, 1 = +bias, 2 = +bias then GELU(exact), 3 = +bias +residual.
// 128x64 block tile, BK=32, 256 threads = 8 waves; wave w owns M-tile w and
// all 4 N-tiles (4 WMMA per K-step, A-fragment reused).  A and transposed B
// staged K-contiguous in LDS; all global offsets are 32-bit (GVS addressing).
// ---------------------------------------------------------------------------
__global__ __launch_bounds__(256) void wmma_gemm_kernel(
    const float* __restrict__ A, const float* __restrict__ W,
    const float* __restrict__ bias, const float* __restrict__ res,
    float* __restrict__ out, int M, int N, int K, int mode)
{
    __shared__ __attribute__((aligned(16))) _Float16 As[128][40]; // [m][k]
    __shared__ __attribute__((aligned(16))) _Float16 Bt[64][40];  // [n][k]

    int tid  = threadIdx.x;
    int m0   = blockIdx.y * 128;
    int n0   = blockIdx.x * 64;
    int wave = tid >> 5;       // M tile 0..7
    int lane = tid & 31;
    int lm   = lane & 15;
    int hi   = lane >> 4;
    int base = hi * 8;

    // loader roles
    int arow = tid >> 1;            // 0..127
    int akse = (tid & 1) << 4;      // 0,16
    int bn   = tid & 63;            // 0..63
    int bkse = (tid >> 6) << 3;     // 0,8,16,24
    int gn   = n0 + bn;
    unsigned uN = (unsigned)N, uK = (unsigned)K;

    FragC acc[4];
    #pragma unroll
    for (int t = 0; t < 4; ++t)
        #pragma unroll
        for (int r = 0; r < 8; ++r) acc[t].f[r] = 0.f;

    for (int k0 = 0; k0 < K; k0 += 32) {
        __syncthreads();
        // A tile: 16 f32 per thread (4x b128 loads -> 2x b128 LDS stores)
        {
            int gm = m0 + arow;
            v8h a0, a1;
            if (gm < M) {
                const v4f* ap = (const v4f*)(A + (unsigned)gm * uK + (unsigned)(k0 + akse));
                a0 = pack8(ap[0], ap[1]);
                a1 = pack8(ap[2], ap[3]);
            } else {
                a0 = zero8(); a1 = zero8();
            }
            *(v8h*)&As[arow][akse]     = a0;
            *(v8h*)&As[arow][akse + 8] = a1;
        }
        // B tile (transposed into LDS): 8 coalesced b32 loads -> one b128 store
        {
            Half8 bvv;
            if (gn < N) {
                unsigned off = (unsigned)(k0 + bkse) * uN + (unsigned)gn;
                #pragma unroll
                for (int e = 0; e < 8; ++e)
                    bvv.h[e] = (_Float16)W[off + (unsigned)e * uN];
            } else {
                bvv.v = zero8();
            }
            *(v8h*)&Bt[bn][bkse] = bvv.v;
        }
        // prefetch next K tile of A
        if (k0 + 32 < K) {
            int gm = m0 + arow;
            if (gm < M) __builtin_prefetch(A + (unsigned)gm * uK + (unsigned)(k0 + 32 + akse), 0, 1);
        }
        __syncthreads();

        v16h fa = frag16(&As[wave * 16 + lm][0], base);
        #pragma unroll
        for (int t = 0; t < 4; ++t) {
            v16h fb = frag16(&Bt[t * 16 + lm][0], base);
            acc[t].v = __builtin_amdgcn_wmma_f32_16x16x32_f16(
                false, fa, false, fb, (short)0, acc[t].v, false, false);
        }
    }

    // epilogue
    #pragma unroll
    for (int t = 0; t < 4; ++t) {
        #pragma unroll
        for (int r = 0; r < 8; ++r) {
            int gm  = m0 + wave * 16 + r + hi * 8;
            int gnn = n0 + t * 16 + lm;
            if (gm < M && gnn < N) {
                float v = acc[t].f[r];
                if (mode >= 1) v += bias[gnn];
                if (mode == 2) v = 0.5f * v * (1.0f + erff(v * 0.70710678118654752f));
                unsigned oidx = (unsigned)gm * uN + (unsigned)gnn;
                if (mode == 3) v += res[oidx];
                out[oidx] = v;
            }
        }
    }
}

// ---------------------------------------------------------------------------
// Windowed attention: one block per (batch, window, head).  49 tokens padded
// to 64.  S = QK^T (WMMA, K=32), +rel-pos bias (+shift mask), softmax,
// O = P V (WMMA, K=64).  Cyclic shift folded into gather/scatter indices.
// qs/ks/ps are K-contiguous; V is stored transposed (vst[d][token]) so the
// P.V B-fragment is also K-contiguous.
// ---------------------------------------------------------------------------
__global__ __launch_bounds__(256) void window_attn_kernel(
    const float* __restrict__ qkv, const float* __restrict__ rpb,
    float* __restrict__ outO, int Hc, int Wc, int C, int heads, int shift)
{
    __shared__ __attribute__((aligned(16))) _Float16 qs[64][40];   // [m][d]
    __shared__ __attribute__((aligned(16))) _Float16 ks[64][40];   // [n][d]
    __shared__ __attribute__((aligned(16))) _Float16 vst[32][72];  // [d][token]
    __shared__ __attribute__((aligned(16))) _Float16 ps[64][72];   // [m][j]
    __shared__ float Sm[64][65];
    __shared__ int   rowmap[64];
    __shared__ int   region[64];

    int tid  = threadIdx.x;
    int head = blockIdx.x % heads;
    int wg   = blockIdx.x / heads;
    int nw   = Wc / 7;
    int nh   = Hc / 7;
    int wx   = wg % nw;
    int t1   = wg / nw;
    int wy   = t1 % nh;
    int b    = t1 / nh;

    // phase 0: token -> global-row map (roll folded in) + mask region ids
    if (tid < 64) {
        if (tid < 49) {
            int ty = tid / 7, tx = tid % 7;
            int hr = wy * 7 + ty, wr = wx * 7 + tx;   // coords in rolled image
            int h = hr + shift; if (h >= Hc) h -= Hc;
            int w = wr + shift; if (w >= Wc) w -= Wc;
            rowmap[tid] = (b * Hc + h) * Wc + w;
            int rh = (hr < Hc - 7) ? 0 : ((hr < Hc - 3) ? 1 : 2);
            int rw = (wr < Wc - 7) ? 0 : ((wr < Wc - 3) ? 1 : 2);
            region[tid] = rh * 3 + rw;
        } else {
            rowmap[tid] = -1;
            region[tid] = -1;
        }
    }
    __syncthreads();

    // phase 1: gather q,k,v; 4 threads per token, 8 d-values each, b128 loads
    {
        unsigned C3 = 3u * (unsigned)C;
        int t  = tid >> 2;            // 0..63
        int ds = (tid & 3) << 3;      // 0,8,16,24
        int row = rowmap[t];
        v8h qv, kv, vv;
        if (row >= 0) {
            const float* qp = qkv + (unsigned)row * C3 + (unsigned)(head * 32 + ds);
            qv = pack8(((const v4f*)qp)[0], ((const v4f*)qp)[1]);
            const float* kp = qp + C;
            kv = pack8(((const v4f*)kp)[0], ((const v4f*)kp)[1]);
            const float* vp = qp + 2 * C;
            vv = pack8(((const v4f*)vp)[0], ((const v4f*)vp)[1]);
        } else {
            qv = zero8(); kv = zero8(); vv = zero8();
        }
        *(v8h*)&qs[t][ds] = qv;
        *(v8h*)&ks[t][ds] = kv;
        Half8 vs8; vs8.v = vv;
        #pragma unroll
        for (int e = 0; e < 8; ++e)
            vst[ds + e][t] = vs8.h[e];
    }
    __syncthreads();

    int lane = tid & 31;
    int lm   = lane & 15;
    int hi   = lane >> 4;
    int base = hi * 8;
    int wave = tid >> 5;
    const float scale = 0.1767766952966369f;  // 1/sqrt(32)

    // phase 2: S = q @ k^T  (16 tiles, 2 per wave)
    {
        int wm = wave & 3;
        int wn = wave >> 2;
        v16h fa = frag16(&qs[wm * 16 + lm][0], base);
        #pragma unroll
        for (int t = 0; t < 2; ++t) {
            int tn = wn * 2 + t;
            v16h fb = frag16(&ks[tn * 16 + lm][0], base);  // B[k][n] = K[n][k]
            FragC acc;
            #pragma unroll
            for (int r = 0; r < 8; ++r) acc.f[r] = 0.f;
            acc.v = __builtin_amdgcn_wmma_f32_16x16x32_f16(
                false, fa, false, fb, (short)0, acc.v, false, false);
            #pragma unroll
            for (int r = 0; r < 8; ++r)
                Sm[wm * 16 + r + hi * 8][tn * 16 + lm] = acc.f[r] * scale;
        }
    }
    __syncthreads();

    // phase 3: +bias(+mask), softmax, store P in f16
    if (tid < 64) {
        int r = tid;
        if (r < 49) {
            int ry = r / 7, rx = r % 7;
            float mx = -1e30f;
            for (int j = 0; j < 49; ++j) {
                int jy = j / 7, jx = j % 7;
                int idx = (ry - jy + 6) * 13 + (rx - jx + 6);
                float s = Sm[r][j] + rpb[idx * heads + head];
                if (shift > 0 && region[r] != region[j]) s += -100.0f;
                Sm[r][j] = s;
                mx = fmaxf(mx, s);
            }
            float sum = 0.f;
            for (int j = 0; j < 49; ++j) {
                float e = expf(Sm[r][j] - mx);
                Sm[r][j] = e;
                sum += e;
            }
            float inv = 1.0f / sum;
            for (int j = 0; j < 64; ++j)
                ps[r][j] = (j < 49) ? (_Float16)(Sm[r][j] * inv) : (_Float16)0.0f;
        } else {
            for (int j = 0; j < 64; ++j) ps[r][j] = (_Float16)0.0f;
        }
    }
    __syncthreads();

    // phase 4: O = P @ v  (64x32, K=64 -> 8 tiles, 1 per wave)
    {
        int wm  = wave & 3;
        int wn2 = wave >> 2;
        FragC acc;
        #pragma unroll
        for (int r = 0; r < 8; ++r) acc.f[r] = 0.f;
        #pragma unroll
        for (int kc = 0; kc < 2; ++kc) {
            v16h fa = frag16(&ps[wm * 16 + lm][kc * 32], base);
            v16h fb = frag16(&vst[wn2 * 16 + lm][kc * 32], base);  // B[k][n]=v[k][n]
            acc.v = __builtin_amdgcn_wmma_f32_16x16x32_f16(
                false, fa, false, fb, (short)0, acc.v, false, false);
        }
        #pragma unroll
        for (int r = 0; r < 8; ++r) {
            int t = wm * 16 + r + hi * 8;
            int d = wn2 * 16 + lm;
            if (t < 49)
                outO[(unsigned)rowmap[t] * (unsigned)C + (unsigned)(head * 32 + d)] = acc.f[r];
        }
    }
}

// ---------------------------------------------------------------------------
// Patch-merge gather: (B,H,W,C) -> (B*H/2*W/2, 4C) concat [x0,x1,x2,x3].
// ---------------------------------------------------------------------------
__global__ __launch_bounds__(256) void merge_concat_kernel(
    const float* __restrict__ in, float* __restrict__ out,
    int H, int W, int C, int total)
{
    int i = blockIdx.x * 256 + threadIdx.x;
    if (i >= total) return;
    int C4 = 4 * C;
    int c4 = i % C4;
    int t  = i / C4;
    int W2 = W / 2;
    int H2 = H / 2;
    int ow = t % W2; t /= W2;
    int oh = t % H2;
    int b  = t / H2;
    int part = c4 / C;
    int c    = c4 % C;
    // x0: (even,even)  x1: (odd,even)  x2: (even,odd)  x3: (odd,odd)
    int dh = (part == 1 || part == 3) ? 1 : 0;
    int dw = (part >= 2) ? 1 : 0;
    out[i] = in[(unsigned)((b * H + 2 * oh + dh) * W + 2 * ow + dw) * (unsigned)C + (unsigned)c];
}

// ---------------------------------------------------------------------------
// Global average pool over the 7x7 final grid.
// ---------------------------------------------------------------------------
__global__ __launch_bounds__(256) void pool_kernel(
    const float* __restrict__ in, float* __restrict__ out)
{
    int i = blockIdx.x * 256 + threadIdx.x;
    if (i >= 32 * 768) return;
    int c = i % 768;
    int b = i / 768;
    float s = 0.f;
    for (int t = 0; t < 49; ++t)
        s += in[(unsigned)(b * 49 + t) * 768u + (unsigned)c];
    out[i] = s * (1.0f / 49.0f);
}

// ---------------------------------------------------------------------------
// Host orchestration.
// ---------------------------------------------------------------------------
struct BlockP {
    const float *ln1_b, *ln1_g, *ln2_b, *ln2_g;
    const float *mlp_b1, *mlp_b2, *mlp_w1, *mlp_w2;
    const float *proj_b, *proj_w, *qkv_b, *qkv_w, *rpb;
};

extern "C" void kernel_launch(void* const* d_in, const int* in_sizes, int n_in,
                              void* d_out, int out_size, void* d_ws, size_t ws_size,
                              hipStream_t stream)
{
    (void)in_sizes; (void)n_in; (void)out_size; (void)ws_size;
    static const int DEPTHS[4] = {2, 2, 6, 2};
    static const int HEADS[4]  = {3, 6, 12, 24};

    // d_in[0] = x (dict order); params flattened JAX-style (alphabetical keys).
    const float* x_in = (const float*)d_in[0];
    int p = 1;
    const float* head_b = (const float*)d_in[p++];
    const float* head_w = (const float*)d_in[p++];
    const float *merge_ln_b[3], *merge_ln_g[3], *merge_w[3];
    for (int i = 0; i < 3; ++i) {
        merge_ln_b[i] = (const float*)d_in[p++];
        merge_ln_g[i] = (const float*)d_in[p++];
        merge_w[i]    = (const float*)d_in[p++];
    }
    const float* norm_b = (const float*)d_in[p++];
    const float* norm_g = (const float*)d_in[p++];
    const float* patch_b    = (const float*)d_in[p++];
    const float* patch_ln_b = (const float*)d_in[p++];
    const float* patch_ln_g = (const float*)d_in[p++];
    const float* patch_w    = (const float*)d_in[p++];
    BlockP blk[12];
    int bi = 0;
    for (int s = 0; s < 4; ++s)
        for (int d = 0; d < DEPTHS[s]; ++d, ++bi) {
            BlockP& q = blk[bi];
            q.ln1_b  = (const float*)d_in[p++]; q.ln1_g  = (const float*)d_in[p++];
            q.ln2_b  = (const float*)d_in[p++]; q.ln2_g  = (const float*)d_in[p++];
            q.mlp_b1 = (const float*)d_in[p++]; q.mlp_b2 = (const float*)d_in[p++];
            q.mlp_w1 = (const float*)d_in[p++]; q.mlp_w2 = (const float*)d_in[p++];
            q.proj_b = (const float*)d_in[p++]; q.proj_w = (const float*)d_in[p++];
            q.qkv_b  = (const float*)d_in[p++]; q.qkv_w  = (const float*)d_in[p++];
            q.rpb    = (const float*)d_in[p++];
        }

    // Workspace carve-up (all f32):
    const size_t nTok = 100352UL * 96UL;   // 9,633,792 floats: max M*C
    float* buf_x   = (float*)d_ws;         // residual stream
    float* buf_a   = buf_x + nTok;         // LN output
    float* buf_o   = buf_a + nTok;         // attn-out / concat / pooled
    float* buf_big = buf_o + nTok;         // qkv (M*3C) / mlp hidden (M*4C)

    // Patch embed + LN
    {
        int total = 32 * 56 * 56 * 96;
        patch_embed_kernel<<<(total + 255) / 256, 256, 0, stream>>>(
            x_in, patch_w, patch_b, buf_o, total);
        layernorm_kernel<<<32 * 56 * 56, 256, 0, stream>>>(
            buf_o, patch_ln_g, patch_ln_b, buf_x, 96);
    }

    bi = 0;
    for (int s = 0; s < 4; ++s) {
        int C = 96 << s;
        int H = 56 >> s;
        int M = 32 * H * H;
        int nWin = (H / 7) * (H / 7);
        int heads = HEADS[s];
        dim3 blkDim(256);
        for (int j = 0; j < DEPTHS[s]; ++j, ++bi) {
            const BlockP& q = blk[bi];
            int shift = (j % 2 == 1 && H > 7) ? 3 : 0;
            // LN1
            layernorm_kernel<<<M, blkDim, 0, stream>>>(buf_x, q.ln1_g, q.ln1_b, buf_a, C);
            // QKV GEMM: M x 3C
            {
                dim3 g((3 * C + 63) / 64, (M + 127) / 128);
                wmma_gemm_kernel<<<g, blkDim, 0, stream>>>(
                    buf_a, q.qkv_w, q.qkv_b, nullptr, buf_big, M, 3 * C, C, 1);
            }
            // Window attention
            window_attn_kernel<<<32 * nWin * heads, blkDim, 0, stream>>>(
                buf_big, q.rpb, buf_o, H, H, C, heads, shift);
            // Proj GEMM + residual
            {
                dim3 g((C + 63) / 64, (M + 127) / 128);
                wmma_gemm_kernel<<<g, blkDim, 0, stream>>>(
                    buf_o, q.proj_w, q.proj_b, buf_x, buf_x, M, C, C, 3);
            }
            // LN2
            layernorm_kernel<<<M, blkDim, 0, stream>>>(buf_x, q.ln2_g, q.ln2_b, buf_a, C);
            // MLP1 (GELU)
            {
                dim3 g((4 * C + 63) / 64, (M + 127) / 128);
                wmma_gemm_kernel<<<g, blkDim, 0, stream>>>(
                    buf_a, q.mlp_w1, q.mlp_b1, nullptr, buf_big, M, 4 * C, C, 2);
            }
            // MLP2 + residual
            {
                dim3 g((C + 63) / 64, (M + 127) / 128);
                wmma_gemm_kernel<<<g, blkDim, 0, stream>>>(
                    buf_big, q.mlp_w2, q.mlp_b2, buf_x, buf_x, M, C, 4 * C, 3);
            }
        }
        if (s < 3) {
            int Mn = M / 4;
            int total = Mn * 4 * C;
            merge_concat_kernel<<<(total + 255) / 256, 256, 0, stream>>>(
                buf_x, buf_o, H, H, C, total);
            layernorm_kernel<<<Mn, 256, 0, stream>>>(
                buf_o, merge_ln_g[s], merge_ln_b[s], buf_a, 4 * C);
            dim3 g((2 * C + 63) / 64, (Mn + 127) / 128);
            wmma_gemm_kernel<<<g, 256, 0, stream>>>(
                buf_a, merge_w[s], nullptr, nullptr, buf_x, Mn, 2 * C, 4 * C, 0);
        }
    }

    // Final LN, pool, head GEMM -> d_out (32 x 1000, f32)
    layernorm_kernel<<<1568, 256, 0, stream>>>(buf_x, norm_g, norm_b, buf_a, 768);
    pool_kernel<<<(32 * 768 + 255) / 256, 256, 0, stream>>>(buf_a, buf_o);
    {
        dim3 g((1000 + 63) / 64, (32 + 127) / 128);
        wmma_gemm_kernel<<<g, 256, 0, stream>>>(
            buf_o, head_w, head_b, nullptr, (float*)d_out, 32, 1000, 768, 1);
    }
}
